// MoEMlp_2559800508437
// MI455X (gfx1250) — matrix operand order repaired and checked by reading, not compiled, not run on hardware
//
#include <hip/hip_runtime.h>
#include <hip/hip_bf16.h>

typedef unsigned short u16;
typedef unsigned int u32;

#define T_TOK 8192
#define DMODEL 1024
#define NEXP 8
#define HDIM 4096
#define CAPACITY 2560

typedef __attribute__((ext_vector_type(16))) __bf16 v16bf;
typedef __attribute__((ext_vector_type(8)))  float  v8f;

static __device__ __forceinline__ u16 f2bf(float f) {
    u32 u = __builtin_bit_cast(u32, f);
    u32 r = (u + 0x7fffu + ((u >> 16) & 1u)) >> 16;
    return (u16)r;
}

// WMMA wrapper: reuse flag must be a frontend constant -> template parameter
template <bool RA>
static __device__ __forceinline__ v8f wmma_bf16(v16bf a, v16bf b, v8f c) {
    return __builtin_amdgcn_wmma_f32_16x16x32_bf16(false, a, false, b, (short)0, c,
                                                   /*reuse_a=*/RA, /*reuse_b=*/false);
}

// ---------------------------------------------------------------- utilities
__global__ void zero_u32_kernel(u32* p, size_t n) {
    for (size_t i = (size_t)blockIdx.x * 256 + threadIdx.x; i < n; i += (size_t)gridDim.x * 256)
        p[i] = 0u;
}

__global__ void fill_i32_kernel(int* p, int v, size_t n) {
    for (size_t i = (size_t)blockIdx.x * 256 + threadIdx.x; i < n; i += (size_t)gridDim.x * 256)
        p[i] = v;
}

// ------------------------------------------------- pack weights to B-fragment layout
// dst layout: [e][ntile][kblock][lane][16 bf16]
// B fragment (16x16x32 bf16): lane L -> column n = L%16, half = L/16,
// element j -> k = kblock*32 + half*8 + (j<8 ? j : j+8)   (ISA table order)
__global__ __launch_bounds__(256) void pack_kernel(const float* __restrict__ w, u16* __restrict__ wp,
                                                   int Kdim, int Ndim) {
    const int NT = Ndim >> 4, KB = Kdim >> 5;
    size_t id = (size_t)blockIdx.x * 256 + threadIdx.x;
    size_t total = (size_t)NEXP * NT * KB * 32;
    if (id >= total) return;
    int lane = (int)(id & 31);
    size_t r = id >> 5;
    int kb = (int)(r % KB); r /= KB;
    int nt = (int)(r % NT);
    int e  = (int)(r / NT);
    int half = lane >> 4;
    int n = nt * 16 + (lane & 15);
    int k0 = kb * 32 + half * 8;
    const float* src = w + ((size_t)e * Kdim + k0) * Ndim + n;
    u16* dst = wp + id * 16;
#pragma unroll
    for (int j = 0; j < 16; ++j) dst[j] = f2bf(src[(size_t)(j < 8 ? j : j + 8) * Ndim]);
}

// ------------------------------------------------------------------ gating
// one wave per token: logits = x[t,:] @ wg, softmax, top-2, normalize
__global__ __launch_bounds__(256) void gate_kernel(const float* __restrict__ x,
                                                   const float* __restrict__ wg,
                                                   int* __restrict__ topi, float* __restrict__ topw,
                                                   float* __restrict__ mece) {
    const int lane = threadIdx.x & 31;
    const int t = blockIdx.x * 8 + (threadIdx.x >> 5);
    float p[NEXP] = {};
    const float* xr = x + (size_t)t * DMODEL;
    for (int d = lane; d < DMODEL; d += 32) {
        float xv = xr[d];
        const float* g = wg + (size_t)d * NEXP;
#pragma unroll
        for (int e = 0; e < NEXP; ++e) p[e] += xv * g[e];
    }
#pragma unroll
    for (int e = 0; e < NEXP; ++e)
        for (int off = 16; off > 0; off >>= 1)
            p[e] += __shfl_xor(p[e], off, 32);
    if (lane == 0) {
        float mx = p[0];
#pragma unroll
        for (int e = 1; e < NEXP; ++e) mx = fmaxf(mx, p[e]);
        float g[NEXP], s = 0.f;
#pragma unroll
        for (int e = 0; e < NEXP; ++e) { g[e] = __expf(p[e] - mx); s += g[e]; }
        float inv = 1.f / s;
#pragma unroll
        for (int e = 0; e < NEXP; ++e) g[e] *= inv;
        int i1 = 0;
#pragma unroll
        for (int e = 1; e < NEXP; ++e) if (g[e] > g[i1]) i1 = e;
        int i2 = (i1 == 0) ? 1 : 0;
#pragma unroll
        for (int e = 0; e < NEXP; ++e) if (e != i1 && g[e] > g[i2]) i2 = e;
        float v1 = g[i1], v2 = g[i2];
        float denom = v1 + v2 + 1e-9f;
        topi[t * 2 + 0] = i1; topi[t * 2 + 1] = i2;
        topw[t * 2 + 0] = v1 / denom; topw[t * 2 + 1] = v2 / denom;
#pragma unroll
        for (int e = 0; e < NEXP; ++e) atomicAdd(&mece[e], g[e]);
        atomicAdd(&mece[NEXP + i1], 1.0f);
    }
}

// ------------------------------------------------ slot-major capacity assignment
// single block, 16 waves: wave w -> (slot = w/8, expert = w%8)
__global__ __launch_bounds__(512) void assign_kernel(const int* __restrict__ topi,
                                                     const float* __restrict__ topw,
                                                     int* __restrict__ rowToken,
                                                     float* __restrict__ rowWeight) {
    const int lane = threadIdx.x & 31;
    const int wv = threadIdx.x >> 5;
    const int e = wv & 7, s = wv >> 3;
    int running = 0;
    if (s == 1) {
        for (int t0 = 0; t0 < T_TOK; t0 += 32) {
            bool m = (topi[(t0 + lane) * 2 + 0] == e);
            running += __popc((unsigned)__ballot(m));
        }
    }
    for (int t0 = 0; t0 < T_TOK; t0 += 32) {
        int t = t0 + lane;
        bool m = (topi[t * 2 + s] == e);
        unsigned mask = (unsigned)__ballot(m);
        if (m) {
            int pos = running + __popc(mask & ((1u << lane) - 1u));
            if (pos < CAPACITY) {
                rowToken[e * CAPACITY + pos] = t;
                rowWeight[e * CAPACITY + pos] = topw[t * 2 + s];
            }
        }
        running += __popc(mask);
    }
}

// ------------------------------------------------------------------ dispatch
__global__ __launch_bounds__(128) void dispatch_kernel(const float* __restrict__ x,
                                                       const int* __restrict__ rowToken,
                                                       u16* __restrict__ disp) {
    int r = blockIdx.x;
    int t = rowToken[r];
    if (t < 0) return;
    const float* src = x + (size_t)t * DMODEL;
    u16* dst = disp + (size_t)r * DMODEL;
    for (int i = threadIdx.x; i < DMODEL; i += 128) dst[i] = f2bf(src[i]);
}

// --------------------------------------------------------------- fused expert FFN
// per workgroup: 16 rows of one expert; fc1 -> gelu -> fc2 -> gated atomic combine
#define KB1 (DMODEL / 32)   // 32
#define NT1 (HDIM / 16)     // 256
#define KB2 (HDIM / 32)     // 128
#define NT2 (DMODEL / 16)   // 64
#define PFD 16              // B-stream prefetch distance in k-blocks (16 KB ahead)

__global__ __launch_bounds__(256) void ffn_kernel(const u16* __restrict__ disp,
                                                  const u16* __restrict__ w1p,
                                                  const u16* __restrict__ w2p,
                                                  const float* __restrict__ b1,
                                                  const float* __restrict__ b2,
                                                  const int* __restrict__ rowToken,
                                                  const float* __restrict__ rowWeight,
                                                  float* __restrict__ y) {
    __shared__ u16 ldsA[KB1 * 32 * 16];   // 32 KB : A panel, fragment layout [kb][lane][16]
    __shared__ u16 ldsH[KB2 * 32 * 16];   // 128 KB: h panel, fragment layout [kb][lane][16]

    const int e = blockIdx.x / (CAPACITY / 16);
    const int rb = blockIdx.x % (CAPACITY / 16);
    const int rowBase = e * CAPACITY + rb * 16;
    const int lane = threadIdx.x & 31;
    const int wave = threadIdx.x >> 5;
    const int half = lane >> 4;

    // ---- stage A panel (disp rows, bf16) into ISA A-fragment layout:
    // lane l (m=l%16, half=l/16), element j -> k = (j<8 ? j : j+8) + half*8  (+ kb*32)
    for (int task = threadIdx.x; task < KB1 * 32; task += 256) {
        int kb = task >> 5, l = task & 31;
        int m = l & 15, hf = l >> 4;
        const u16* src = disp + (size_t)(rowBase + m) * DMODEL + kb * 32 + hf * 8;
        uint4 run0 = *(const uint4*)src;          // j = 0..7
        uint4 run1 = *(const uint4*)(src + 16);   // j = 8..15 (k+16)
        uint4* dst = (uint4*)&ldsA[task * 16];
        dst[0] = run0; dst[1] = run1;
    }
    __syncthreads();

    // ---- fc1: wave handles H columns [wave*512, wave*512+512), 4 tiles in flight
#pragma unroll 1
    for (int g = 0; g < 8; ++g) {
        int nt0 = wave * 32 + g * 4;
        v8f acc0 = {}, acc1 = {}, acc2 = {}, acc3 = {};
#pragma unroll 1
        for (int kb = 0; kb < KB1; ++kb) {
            v16bf a = *(const v16bf*)&ldsA[(kb * 32 + lane) * 16];
            const u16* bp = w1p + ((((size_t)e * NT1 + nt0) * KB1 + kb) * 32 + lane) * 16;
            const size_t ntStride = (size_t)KB1 * 32 * 16;   // one ntile = KB1 k-blocks
            __builtin_prefetch(bp + 0 * ntStride + (size_t)PFD * 32 * 16, 0, 1);
            __builtin_prefetch(bp + 1 * ntStride + (size_t)PFD * 32 * 16, 0, 1);
            __builtin_prefetch(bp + 2 * ntStride + (size_t)PFD * 32 * 16, 0, 1);
            __builtin_prefetch(bp + 3 * ntStride + (size_t)PFD * 32 * 16, 0, 1);
            v16bf b0 = *(const v16bf*)(bp + 0 * ntStride);
            v16bf b1f = *(const v16bf*)(bp + 1 * ntStride);
            v16bf b2f = *(const v16bf*)(bp + 2 * ntStride);
            v16bf b3 = *(const v16bf*)(bp + 3 * ntStride);
            // same A operand for all 4 WMMAs of this kb -> reuse_a hint after the first
            acc0 = wmma_bf16<false>(a, b0, acc0);
            acc1 = wmma_bf16<true >(a, b1f, acc1);
            acc2 = wmma_bf16<true >(a, b2f, acc2);
            acc3 = wmma_bf16<true >(a, b3, acc3);
        }
        // bias + exact GELU, store h in A-fragment layout for fc2
        v8f accs[4] = {acc0, acc1, acc2, acc3};
#pragma unroll
        for (int q = 0; q < 4; ++q) {
            int n = (nt0 + q) * 16 + (lane & 15);
            float bias = b1[e * HDIM + n];
            int kb2 = n >> 5, kk = n & 31;
            int half2 = (kk >> 3) & 1;
            int j = (kk & 7) + (kk >= 16 ? 8 : 0);
#pragma unroll
            for (int r = 0; r < 8; ++r) {
                int m = r + half * 8;
                float v = accs[q][r] + bias;
                v = 0.5f * v * (1.0f + erff(v * 0.70710678118f));
                ldsH[((kb2 * 32) + (m + 16 * half2)) * 16 + j] = f2bf(v);
            }
        }
    }
    __syncthreads();

    // ---- fc2: wave handles D columns [wave*128, wave*128+128)
#pragma unroll 1
    for (int g = 0; g < 2; ++g) {
        int nt0 = wave * 8 + g * 4;
        v8f acc0 = {}, acc1 = {}, acc2 = {}, acc3 = {};
#pragma unroll 1
        for (int kb = 0; kb < KB2; ++kb) {
            v16bf a = *(const v16bf*)&ldsH[(kb * 32 + lane) * 16];
            const u16* bp = w2p + ((((size_t)e * NT2 + nt0) * KB2 + kb) * 32 + lane) * 16;
            const size_t ntStride = (size_t)KB2 * 32 * 16;
            __builtin_prefetch(bp + 0 * ntStride + (size_t)PFD * 32 * 16, 0, 1);
            __builtin_prefetch(bp + 1 * ntStride + (size_t)PFD * 32 * 16, 0, 1);
            __builtin_prefetch(bp + 2 * ntStride + (size_t)PFD * 32 * 16, 0, 1);
            __builtin_prefetch(bp + 3 * ntStride + (size_t)PFD * 32 * 16, 0, 1);
            v16bf b0 = *(const v16bf*)(bp + 0 * ntStride);
            v16bf b1f = *(const v16bf*)(bp + 1 * ntStride);
            v16bf b2f = *(const v16bf*)(bp + 2 * ntStride);
            v16bf b3 = *(const v16bf*)(bp + 3 * ntStride);
            acc0 = wmma_bf16<false>(a, b0, acc0);
            acc1 = wmma_bf16<true >(a, b1f, acc1);
            acc2 = wmma_bf16<true >(a, b2f, acc2);
            acc3 = wmma_bf16<true >(a, b3, acc3);
        }
        // gated combine: y[t] += gate * (val + b2)
        v8f accs[4] = {acc0, acc1, acc2, acc3};
#pragma unroll
        for (int q = 0; q < 4; ++q) {
            int n = (nt0 + q) * 16 + (lane & 15);
            float bias = b2[e * DMODEL + n];
#pragma unroll
            for (int r = 0; r < 8; ++r) {
                int m = r + half * 8;
                int row = rowBase + m;
                int t = rowToken[row];
                if (t >= 0) {
                    float wgt = rowWeight[row];
                    atomicAdd(&y[(size_t)t * DMODEL + n], wgt * (accs[q][r] + bias));
                }
            }
        }
    }
}

// ------------------------------------------------------------------ l_aux
__global__ void laux_kernel(const float* __restrict__ mece, float* __restrict__ out) {
    float s = 0.f;
    for (int e = 0; e < NEXP; ++e) s += mece[e] * mece[NEXP + e];
    out[0] = (float)NEXP * s / ((float)T_TOK * (float)T_TOK);
}

// =====================================================================
extern "C" void kernel_launch(void* const* d_in, const int* in_sizes, int n_in,
                              void* d_out, int out_size, void* d_ws, size_t ws_size,
                              hipStream_t stream) {
    const float* x  = (const float*)d_in[0];   // [T, D]
    const float* wg = (const float*)d_in[1];   // [D, E]
    const float* w1 = (const float*)d_in[2];   // [E, D, H]
    const float* b1 = (const float*)d_in[3];   // [E, H]
    const float* w2 = (const float*)d_in[4];   // [E, H, D]
    const float* b2 = (const float*)d_in[5];   // [E, D]
    float* y = (float*)d_out;                  // [T*D] + 1 (l_aux)

    char* ws = (char*)d_ws;
    const size_t w1p_bytes  = (size_t)NEXP * DMODEL * HDIM * 2;        // 64 MB
    const size_t w2p_bytes  = (size_t)NEXP * HDIM * DMODEL * 2;        // 64 MB
    const size_t disp_bytes = (size_t)NEXP * CAPACITY * DMODEL * 2;    // 40 MB
    u16*   w1p  = (u16*)(ws);
    u16*   w2p  = (u16*)(ws + w1p_bytes);
    u16*   disp = (u16*)(ws + w1p_bytes + w2p_bytes);
    char*  p    = ws + w1p_bytes + w2p_bytes + disp_bytes;
    int*   topi      = (int*)p;                 p += (size_t)T_TOK * 2 * 4;
    float* topw      = (float*)p;               p += (size_t)T_TOK * 2 * 4;
    int*   rowToken  = (int*)p;                 p += (size_t)NEXP * CAPACITY * 4;
    float* rowWeight = (float*)p;               p += (size_t)NEXP * CAPACITY * 4;
    float* mece      = (float*)p;               // 16 floats

    // 0) init scratch + output accumulators
    zero_u32_kernel<<<4096, 256, 0, stream>>>((u32*)disp, disp_bytes / 4);
    zero_u32_kernel<<<4096, 256, 0, stream>>>((u32*)y, (size_t)T_TOK * DMODEL);
    zero_u32_kernel<<<64, 256, 0, stream>>>((u32*)rowWeight, (size_t)NEXP * CAPACITY);
    zero_u32_kernel<<<1, 256, 0, stream>>>((u32*)mece, 16);
    fill_i32_kernel<<<64, 256, 0, stream>>>(rowToken, -1, (size_t)NEXP * CAPACITY);

    // 1) pack weights to bf16 WMMA B-fragment layout
    {
        size_t tot1 = (size_t)NEXP * NT1 * KB1 * 32;
        pack_kernel<<<(unsigned)((tot1 + 255) / 256), 256, 0, stream>>>(w1, w1p, DMODEL, HDIM);
        size_t tot2 = (size_t)NEXP * NT2 * KB2 * 32;
        pack_kernel<<<(unsigned)((tot2 + 255) / 256), 256, 0, stream>>>(w2, w2p, HDIM, DMODEL);
    }

    // 2) gating (one wave per token)
    gate_kernel<<<T_TOK / 8, 256, 0, stream>>>(x, wg, topi, topw, mece);

    // 3) capacity assignment (slot-major cumsum)
    assign_kernel<<<1, 512, 0, stream>>>(topi, topw, rowToken, rowWeight);

    // 4) dispatch tokens into expert buffers (bf16)
    dispatch_kernel<<<NEXP * CAPACITY, 128, 0, stream>>>(x, rowToken, disp);

    // 5) fused expert FFN with WMMA + gated combine
    ffn_kernel<<<NEXP * (CAPACITY / 16), 256, 0, stream>>>(disp, w1p, w2p, b1, b2,
                                                           rowToken, rowWeight, y);

    // 6) aux loss
    laux_kernel<<<1, 1, 0, stream>>>(mece, y + (size_t)T_TOK * DMODEL);
}
